// UniversalDirectionalGSA_67688684585066
// MI455X (gfx1250) — matrix operand order repaired and checked by reading, not compiled, not run on hardware
//
#include <hip/hip_runtime.h>
#include <hip/hip_bf16.h>

// ---------------------------------------------------------------------------
// MI455X (gfx1250) fused UniversalDirectionalGSA attention.
// wave32, bf16 WMMA everywhere; TDM (tensor_load_to_lds) double-buffered
// K/V tile prefetch in the attention kernel.
//   ws layout (bf16): Qaug[16][2048][160] | Kaug[16][2048][160]
//                     Vt[16][128][2048]   | Obuf[2048][2048]   (~37.8 MB)
// ---------------------------------------------------------------------------

typedef __attribute__((ext_vector_type(16))) __bf16 v16bf;
typedef __attribute__((ext_vector_type(8)))  __bf16 v8bf;
typedef __attribute__((ext_vector_type(8)))  float  v8f;
typedef __attribute__((ext_vector_type(4)))  unsigned int u32x4;
typedef __attribute__((ext_vector_type(8)))  int i32x8;
typedef __attribute__((ext_vector_type(4)))  int i32x4;

#define SEQ   2048
#define DM    2048
#define NHEAD 16
#define HD    128
#define NSP   8
#define KAUGW 160        // 128 dims + 16 splat features + 16 zero pad
#define KB    32         // key block (2 score subtiles -> one K=32 PV WMMA)
#define NEGINF (-3.0e38f)

#ifdef __AMDGCN__
#if __has_builtin(__builtin_amdgcn_tensor_load_to_lds)
#define HAVE_TDM 1
#endif
#endif
#ifndef HAVE_TDM
#define HAVE_TDM 0
#endif

__device__ __forceinline__ v8f zero8() {
  v8f z;
#pragma unroll
  for (int i = 0; i < 8; ++i) z[i] = 0.0f;
  return z;
}

__device__ __forceinline__ v16bf join16(v8bf lo, v8bf hi) {
  v16bf r;
#pragma unroll
  for (int i = 0; i < 8; ++i) { r[i] = lo[i]; r[i + 8] = hi[i]; }
  return r;
}

// 8 consecutive fp32 -> v8bf (lowers to v_cvt_pk_bf16_f32)
__device__ __forceinline__ v8bf cvt8f(const float* p) {
  float4 a = *(const float4*)p;
  float4 b = *(const float4*)(p + 4);
  v8bf r;
  r[0] = (__bf16)a.x; r[1] = (__bf16)a.y; r[2] = (__bf16)a.z; r[3] = (__bf16)a.w;
  r[4] = (__bf16)b.x; r[5] = (__bf16)b.y; r[6] = (__bf16)b.z; r[7] = (__bf16)b.w;
  return r;
}

__device__ __forceinline__ v8f wmma_bf16(v16bf a, v16bf b, v8f c) {
  // D = A(16x32 bf16) * B(32x16 bf16) + C(16x16 f32)
  return __builtin_amdgcn_wmma_f32_16x16x32_bf16(false, a, false, b,
                                                 (short)0, c, false, false);
}

// row reductions: WMMA C rows live across the 16 lanes of a half-wave
__device__ __forceinline__ float redmax16(float v) {
#pragma unroll
  for (int m = 1; m < 16; m <<= 1) v = fmaxf(v, __shfl_xor(v, m, 32));
  return v;
}
__device__ __forceinline__ float redsum16(float v) {
#pragma unroll
  for (int m = 1; m < 16; m <<= 1) v += __shfl_xor(v, m, 32);
  return v;
}
__device__ __forceinline__ float redsum32(float v) {
#pragma unroll
  for (int m = 1; m < 32; m <<= 1) v += __shfl_xor(v, m, 32);
  return v;
}

#if HAVE_TDM
// Issue one 2-D TDM tile load: tile (tw x th) of 2-byte elements from a
// row-major tensor (row stride = stride elems) into contiguous LDS.
// Descriptor layout per CDNA5 ISA ch.8 (D# group0/group1).
__device__ __forceinline__ void tdm_load2d(unsigned lds_off, const void* g,
                                           unsigned tw, unsigned th,
                                           unsigned td0, unsigned td1,
                                           unsigned stride) {
  unsigned long long ga = (unsigned long long)g;
  u32x4 g0;
  g0[0] = 1u;                                      // count=1, user descriptor
  g0[1] = lds_off;                                 // LDS byte address
  g0[2] = (unsigned)ga;                            // global_addr[31:0]
  g0[3] = (unsigned)((ga >> 32) & 0x1ffffffu) | 0x80000000u;  // [56:32]|type=2
  i32x8 g1;
  g1[0] = 1 << 16;                                 // data_size=1 -> 2 bytes
  g1[1] = (int)((td0 & 0xffffu) << 16);            // tensor_dim0[15:0]
  g1[2] = (int)((td0 >> 16) | ((td1 & 0xffffu) << 16));
  g1[3] = (int)((td1 >> 16) | (tw << 16));         // tile_dim0
  g1[4] = (int)th;                                 // tile_dim1 (tile_dim2=0)
  g1[5] = (int)stride;                             // tensor_dim0_stride lo32
  g1[6] = 0;
  g1[7] = 0;
  i32x4 z4 = {0, 0, 0, 0};
#if __clang_major__ >= 23
  i32x8 z8 = {0, 0, 0, 0, 0, 0, 0, 0};
  __builtin_amdgcn_tensor_load_to_lds(g0, g1, z4, z4, z8, 0);
#else
  __builtin_amdgcn_tensor_load_to_lds(g0, g1, z4, z4, 0);
#endif
}
#endif

// ---------------------------------------------------------------------------
// Kernel 1: QKV projection. out[s,d] = sum_k X[s,k]*W[d,k]  (A @ W^T, "NT")
// grid (DM/64, SEQ/64, 3), block (32,4): each wave owns a 32x32 tile
// (2x2 WMMA accumulators -> 2x the L2 reuse of a 16x16 tile).
// ---------------------------------------------------------------------------
__global__ __launch_bounds__(128) void qkv_kernel(
    const float* __restrict__ X, const float* __restrict__ Wq,
    const float* __restrict__ Wk, const float* __restrict__ Wv,
    __bf16* __restrict__ Qaug, __bf16* __restrict__ Kaug,
    __bf16* __restrict__ Vt) {
  const int z = blockIdx.z;
  const float* W = (z == 0) ? Wq : (z == 1) ? Wk : Wv;
  const int lane = threadIdx.x, Lm = lane & 15, half = lane >> 4;
  const int wave = threadIdx.y;
  const int stile = blockIdx.y * 64 + (wave >> 1) * 32;
  const int dtile = blockIdx.x * 64 + (wave & 1) * 32;

  const float* arow[2] = { X + (size_t)(stile + Lm) * DM,
                           X + (size_t)(stile + 16 + Lm) * DM };
  const float* wrow[2] = { W + (size_t)(dtile + Lm) * DM,
                           W + (size_t)(dtile + 16 + Lm) * DM };

  v8f acc[2][2];
#pragma unroll
  for (int i = 0; i < 2; ++i)
#pragma unroll
    for (int j = 0; j < 2; ++j) acc[i][j] = zero8();

  for (int kb = 0; kb < DM; kb += 32) {
    v16bf af[2], bfr[2];
#pragma unroll
    for (int i = 0; i < 2; ++i) {
      af[i] = join16(cvt8f(arow[i] + kb + half * 8),
                     cvt8f(arow[i] + kb + 16 + half * 8));
      bfr[i] = join16(cvt8f(wrow[i] + kb + half * 8),
                      cvt8f(wrow[i] + kb + 16 + half * 8));
    }
#pragma unroll
    for (int i = 0; i < 2; ++i)
#pragma unroll
      for (int j = 0; j < 2; ++j)
        acc[i][j] = wmma_bf16(af[i], bfr[j], acc[i][j]);
  }

#pragma unroll
  for (int i = 0; i < 2; ++i) {
#pragma unroll
    for (int j = 0; j < 2; ++j) {
      const int d = dtile + j * 16 + Lm, hh = d >> 7, c = d & 127;
#pragma unroll
      for (int r = 0; r < 8; ++r) {
        const int row = stile + i * 16 + half * 8 + r;   // C: M = 8*half + r
        const float v = acc[i][j][r];
        if (z == 0)      Qaug[((size_t)hh * SEQ + row) * KAUGW + c] = (__bf16)v;
        else if (z == 1) Kaug[((size_t)hh * SEQ + row) * KAUGW + c] = (__bf16)v;
        else             Vt[((size_t)hh * HD + c) * SEQ + row]      = (__bf16)v;
      }
    }
  }
}

// ---------------------------------------------------------------------------
// Kernel 2: splat features -> augmented columns 128..159, and fold 1/sqrt(hd)
// into Q.  One wave per (s,h); gridDim.y selects Q vs K side.
// ---------------------------------------------------------------------------
__global__ __launch_bounds__(128) void splat_kernel(
    __bf16* __restrict__ Qaug, __bf16* __restrict__ Kaug,
    const float* __restrict__ pos, const float* __restrict__ dir,
    const float* __restrict__ lsc, const float* __restrict__ lam,
    const float* __restrict__ dstr_p, const float* __restrict__ ps_p) {
  const int lane = threadIdx.x;
  const int pair = blockIdx.x * 4 + threadIdx.y;   // 0..SEQ*NHEAD-1
  const int h = pair & (NHEAD - 1);
  const int s = pair >> 4;
  const int isK = blockIdx.y;
  __bf16* base = (isK ? Kaug : Qaug) + ((size_t)h * SEQ + s) * KAUGW;

  float x[4];
#pragma unroll
  for (int j = 0; j < 4; ++j) x[j] = (float)base[lane + 32 * j];
  const float ps = ps_p[0], dstr = dstr_p[0];

#pragma unroll
  for (int n = 0; n < NSP; ++n) {
    const float* pp = pos + ((size_t)h * NSP + n) * HD;
    const float* dp = dir + ((size_t)h * NSP + n) * HD;
    float dn = 0.0f, dv[4];
#pragma unroll
    for (int j = 0; j < 4; ++j) { dv[j] = dp[lane + 32 * j]; dn += dv[j] * dv[j]; }
    dn = redsum32(dn);
    const float inv = 1.0f / (sqrtf(dn) + 1e-8f);
    float ss = 0.0f, dt = 0.0f;
#pragma unroll
    for (int j = 0; j < 4; ++j) {
      const float df = x[j] - pp[lane + 32 * j];
      ss += df * df;
      dt += df * dv[j] * inv;
    }
    ss = redsum32(ss); dt = redsum32(dt);
    const float scale = __expf(lsc[h * NSP + n]);
    const float amp   = __expf(lam[h * NSP + n]);
    const float g     = __expf(-0.5f * ss / (scale * scale));
    const float dirp  = tanhf(dt);
    if (lane == 0) {
      if (isK) {
        base[128 + 2 * n] = (__bf16)g;
        base[129 + 2 * n] = (__bf16)(g * dirp);
      } else {
        const float a0 = ps * amp * g;
        base[128 + 2 * n] = (__bf16)a0;
        base[129 + 2 * n] = (__bf16)(a0 * dstr * dirp);
      }
    }
  }
  if (lane < 16) base[144 + lane] = (__bf16)0.0f;      // zero pad to K=32
  if (!isK) {
#pragma unroll
    for (int j = 0; j < 4; ++j)
      base[lane + 32 * j] = (__bf16)(x[j] * 0.08838834764831845f); // 1/sqrt(128)
  }
}

// ---------------------------------------------------------------------------
// Kernel 3: fused flash attention with dual online softmax (std + gsa),
// early probability blend, and TDM double-buffered K/V tile prefetch.
// grid (SEQ/64, NHEAD), block (32,4).
// ---------------------------------------------------------------------------
__device__ __forceinline__ void online_update(v8f& s0, v8f& s1, float* m,
                                              float* l, v8f* acc,
                                              __bf16* pls, int Lm, int half) {
#pragma unroll
  for (int r = 0; r < 8; ++r) {
    const float tm = fmaxf(redmax16(s0[r]), redmax16(s1[r]));
    const float nm = fmaxf(m[r], tm);
    const float sc = __expf(m[r] - nm);
    const float p0 = __expf(s0[r] - nm);
    const float p1 = __expf(s1[r] - nm);
    l[r] = l[r] * sc + redsum16(p0) + redsum16(p1);
    m[r] = nm;
#pragma unroll
    for (int f = 0; f < 8; ++f) acc[f][r] *= sc;
    const int row = half * 8 + r;                 // C layout row
    pls[row * KB + Lm]      = (__bf16)p0;
    pls[row * KB + 16 + Lm] = (__bf16)p1;
  }
}

__global__ __launch_bounds__(128) void attn_kernel(
    const __bf16* __restrict__ Qaug, const __bf16* __restrict__ Kaug,
    const __bf16* __restrict__ Vt, __bf16* __restrict__ Obuf,
    const float* __restrict__ gstr_p) {
  const int h = blockIdx.y;
  const int wgq = blockIdx.x * 64;
  const int wave = threadIdx.y;
  const int lane = threadIdx.x, Lm = lane & 15, half = lane >> 4;
  const int qt = wgq + wave * 16;
  const int tid = wave * 32 + lane;
  (void)tid;

  __shared__ __bf16 ldsK[2][KB * KAUGW];   // double-buffered 32 x 160 tile
  __shared__ __bf16 ldsVt[2][HD * KB];     // double-buffered 128 x 32 tile
  __shared__ __bf16 ldsP[4][2][16 * KB];   // per-wave P staging (C -> A layout)

  const float gs = gstr_p[0];
  const float b  = fminf(0.03f, 0.15f / (1.0f + __expf(-gs)));

  // Q A-fragments: 5 chunks of K=32 (4 x Q-dims + 1 x splat features)
  v16bf qf[5];
  {
    const __bf16* qrow = Qaug + ((size_t)h * SEQ + (qt + Lm)) * KAUGW;
#pragma unroll
    for (int c = 0; c < 5; ++c) {
      const __bf16* p0 = qrow + c * 32 + half * 8;
      qf[c] = join16(*(const v8bf*)p0, *(const v8bf*)(p0 + 16));
    }
  }

  v8f acc_s[8], acc_g[8];
  float m_s[8], m_g[8], l_s[8], l_g[8];
#pragma unroll
  for (int f = 0; f < 8; ++f) { acc_s[f] = zero8(); acc_g[f] = zero8(); }
#pragma unroll
  for (int r = 0; r < 8; ++r) { m_s[r] = NEGINF; m_g[r] = NEGINF; l_s[r] = 0.0f; l_g[r] = 0.0f; }

  const int nkb = blockIdx.x * 2 + 2;   // workgroup-uniform: keys < wgq+64

#if HAVE_TDM
  // prologue: async-prefetch tile 0 via the Tensor Data Mover
  if (wave == 0) {
    tdm_load2d((unsigned)(size_t)&ldsK[0][0],
               Kaug + ((size_t)h * SEQ + 0) * KAUGW,
               KAUGW, KB, KAUGW, SEQ, KAUGW);
    tdm_load2d((unsigned)(size_t)&ldsVt[0][0],
               Vt + (size_t)h * HD * SEQ + 0,
               KB, HD, SEQ, HD, SEQ);
  }
#endif

  for (int ib = 0; ib < nkb; ++ib) {
    const int kb = ib * KB;
    const int cur = ib & 1;
    __syncthreads();                    // previous iteration's LDS readers done
#if HAVE_TDM
    if (wave == 0) {
      if (ib + 1 < nkb) {
        // prefetch next tile into the other buffer while this one computes
        const int kn = kb + KB, nx = cur ^ 1;
        tdm_load2d((unsigned)(size_t)&ldsK[nx][0],
                   Kaug + ((size_t)h * SEQ + kn) * KAUGW,
                   KAUGW, KB, KAUGW, SEQ, KAUGW);
        tdm_load2d((unsigned)(size_t)&ldsVt[nx][0],
                   Vt + (size_t)h * HD * SEQ + kn,
                   KB, HD, SEQ, HD, SEQ);
        __builtin_amdgcn_s_wait_tensorcnt((short)2);  // tile ib done (in-order)
      } else {
        __builtin_amdgcn_s_wait_tensorcnt((short)0);
      }
    }
#else
    {   // cooperative K tile load: 16 rows x 160 bf16 = 320 uint4 (half the 32-row tile), contiguous
      const int t2 = wave * 32 + lane;
      const uint4* src = (const uint4*)(Kaug + ((size_t)h * SEQ + kb) * KAUGW);
      uint4* dst = (uint4*)&ldsK[cur][0];
#pragma unroll
      for (int i = 0; i < 3; ++i) {
        const int idx = t2 + i * 128;
        if (idx < 320) dst[idx] = src[idx];
      }
      const uint4* vs = (const uint4*)(Vt + ((size_t)h * HD + t2) * SEQ + kb);
      uint4* vd = (uint4*)(&ldsVt[cur][0] + t2 * KB);
#pragma unroll
      for (int i = 0; i < 4; ++i) vd[i] = vs[i];
    }
#endif
    __syncthreads();

    if (kb <= qt + 15) {                // causal: this block intersects my rows
      const __bf16* ldsKc = &ldsK[cur][0];
      const __bf16* ldsVc = &ldsVt[cur][0];
      v8f s0s, s0g, s1s, s1g;
#pragma unroll
      for (int t = 0; t < 2; ++t) {     // two 16-key subtiles
        const __bf16* krow = ldsKc + (16 * t + Lm) * KAUGW;  // B: lane = key N
        v8f cs = zero8();
#pragma unroll
        for (int c = 0; c < 4; ++c) {   // 128 dims of Q.K^T (scale pre-folded)
          const __bf16* kp = krow + c * 32 + half * 8;
          v16bf kf = join16(*(const v8bf*)kp, *(const v8bf*)(kp + 16));
          cs = wmma_bf16(qf[c], kf, cs);
        }
        // rank-16 splat augmentation -> gsa scores
        const __bf16* kp4 = krow + 128 + half * 8;
        v16bf kf4 = join16(*(const v8bf*)kp4, *(const v8bf*)(kp4 + 16));
        v8f cg = wmma_bf16(qf[4], kf4, cs);
        // causal mask
        const int col = kb + 16 * t + Lm;
#pragma unroll
        for (int r = 0; r < 8; ++r) {
          const int row = qt + half * 8 + r;
          if (col > row) { cs[r] = NEGINF; cg[r] = NEGINF; }
        }
        if (t == 0) { s0s = cs; s0g = cg; } else { s1s = cs; s1g = cg; }
      }

      online_update(s0s, s1s, m_s, l_s, acc_s, &ldsP[wave][0][0], Lm, half);
      online_update(s0g, s1g, m_g, l_g, acc_g, &ldsP[wave][1][0], Lm, half);

      // P (16x32) A-fragments back from wave-private LDS (DS ops in-order)
      const __bf16* ps0 = &ldsP[wave][0][0] + Lm * KB + half * 8;
      const __bf16* pg0 = &ldsP[wave][1][0] + Lm * KB + half * 8;
      v16bf pa_s = join16(*(const v8bf*)ps0, *(const v8bf*)(ps0 + 16));
      v16bf pa_g = join16(*(const v8bf*)pg0, *(const v8bf*)(pg0 + 16));

#pragma unroll
      for (int f = 0; f < 8; ++f) {     // O += P*V over 8 hd chunks
        const __bf16* vr = ldsVc + (16 * f + Lm) * KB + half * 8;
        v16bf vb = join16(*(const v8bf*)vr, *(const v8bf*)(vr + 16));
        acc_s[f] = wmma_bf16(pa_s, vb, acc_s[f]);
        acc_g[f] = wmma_bf16(pa_g, vb, acc_g[f]);
      }
    }
  }

  // normalize + blend + store (heads interleaved: col = h*128 + hd_col)
#pragma unroll
  for (int f = 0; f < 8; ++f) {
#pragma unroll
    for (int r = 0; r < 8; ++r) {
      const int row = qt + half * 8 + r;
      const float val = (1.0f - b) * acc_s[f][r] / l_s[r]
                      + b * acc_g[f][r] / l_g[r];
      Obuf[(size_t)row * DM + h * HD + 16 * f + Lm] = (__bf16)val;
    }
  }
}

// ---------------------------------------------------------------------------
// Kernel 4: final projection  out = O @ Wo^T  (bf16 A, fp32 W, fp32 out)
// 32x32 tile per wave, same structure as qkv_kernel.
// ---------------------------------------------------------------------------
__global__ __launch_bounds__(128) void wo_kernel(
    const __bf16* __restrict__ Obuf, const float* __restrict__ Wo,
    float* __restrict__ out) {
  const int lane = threadIdx.x, Lm = lane & 15, half = lane >> 4;
  const int wave = threadIdx.y;
  const int stile = blockIdx.y * 64 + (wave >> 1) * 32;
  const int dtile = blockIdx.x * 64 + (wave & 1) * 32;

  const __bf16* arow[2] = { Obuf + (size_t)(stile + Lm) * DM,
                            Obuf + (size_t)(stile + 16 + Lm) * DM };
  const float* wrow[2]  = { Wo + (size_t)(dtile + Lm) * DM,
                            Wo + (size_t)(dtile + 16 + Lm) * DM };

  v8f acc[2][2];
#pragma unroll
  for (int i = 0; i < 2; ++i)
#pragma unroll
    for (int j = 0; j < 2; ++j) acc[i][j] = zero8();

  for (int kb = 0; kb < DM; kb += 32) {
    v16bf af[2], bfr[2];
#pragma unroll
    for (int i = 0; i < 2; ++i) {
      const __bf16* ap = arow[i] + kb + half * 8;
      af[i] = join16(*(const v8bf*)ap, *(const v8bf*)(ap + 16));
      bfr[i] = join16(cvt8f(wrow[i] + kb + half * 8),
                      cvt8f(wrow[i] + kb + 16 + half * 8));
    }
#pragma unroll
    for (int i = 0; i < 2; ++i)
#pragma unroll
      for (int j = 0; j < 2; ++j)
        acc[i][j] = wmma_bf16(af[i], bfr[j], acc[i][j]);
  }

#pragma unroll
  for (int i = 0; i < 2; ++i)
#pragma unroll
    for (int j = 0; j < 2; ++j)
#pragma unroll
      for (int r = 0; r < 8; ++r)
        out[(size_t)(stile + i * 16 + half * 8 + r) * DM + dtile + j * 16 + Lm]
            = acc[i][j][r];
}

// ---------------------------------------------------------------------------
extern "C" void kernel_launch(void* const* d_in, const int* in_sizes, int n_in,
                              void* d_out, int out_size, void* d_ws,
                              size_t ws_size, hipStream_t stream) {
  (void)in_sizes; (void)n_in; (void)out_size; (void)ws_size;
  const float* X    = (const float*)d_in[0];
  const float* Wq   = (const float*)d_in[1];
  const float* Wk   = (const float*)d_in[2];
  const float* Wv   = (const float*)d_in[3];
  const float* Wo   = (const float*)d_in[4];
  const float* pos  = (const float*)d_in[5];
  const float* dir  = (const float*)d_in[6];
  const float* lsc  = (const float*)d_in[7];
  const float* lam  = (const float*)d_in[8];
  const float* dstr = (const float*)d_in[9];
  const float* ps   = (const float*)d_in[10];
  const float* gstr = (const float*)d_in[11];

  __bf16* Qaug = (__bf16*)d_ws;
  __bf16* Kaug = Qaug + (size_t)NHEAD * SEQ * KAUGW;
  __bf16* Vt   = Kaug + (size_t)NHEAD * SEQ * KAUGW;
  __bf16* Obuf = Vt   + (size_t)NHEAD * HD * SEQ;

  qkv_kernel<<<dim3(DM / 64, SEQ / 64, 3), dim3(32, 4), 0, stream>>>(
      X, Wq, Wk, Wv, Qaug, Kaug, Vt);
  splat_kernel<<<dim3(SEQ * NHEAD / 4, 2), dim3(32, 4), 0, stream>>>(
      Qaug, Kaug, pos, dir, lsc, lam, dstr, ps);
  attn_kernel<<<dim3(SEQ / 64, NHEAD), dim3(32, 4), 0, stream>>>(
      Qaug, Kaug, Vt, Obuf, gstr);
  wo_kernel<<<dim3(DM / 64, SEQ / 64), dim3(32, 4), 0, stream>>>(
      Obuf, Wo, (float*)d_out);
}
